// Embed_46256797778093
// MI455X (gfx1250) — matrix operand order repaired and checked by reading, not compiled, not run on hardware
//
#include <hip/hip_runtime.h>
#include <hip/hip_bf16.h>
#include <math.h>

typedef __attribute__((ext_vector_type(16))) _Float16 v16h;
typedef __attribute__((ext_vector_type(8)))  _Float16 v8h;
typedef __attribute__((ext_vector_type(8)))  float    v8f;

#define DEV __device__ __forceinline__

constexpr int BG = 64, LLn = 128, EPG = 512;
constexpr int NN = BG * LLn;                 // 8192 nodes
constexpr int IN_DIM = 37, DD = 128, HHn = 8, DKC = 64, EDIM = 6, NLAY = 4;
constexpr int HD = HHn * DKC;                // 512
constexpr int FCK = DKC * (HHn + 1);         // 576

// ---- WMMA fragment helpers (wave32, v_wmma_f32_16x16x32_f16; A/B in this code's contiguous k order) ----
// A (16x32): lanes 0-15 -> M=lane, K 0..15; lanes 16-31 -> M=lane-16, K 16..31
// B (32x16): lanes 0-15 -> N=lane, K 0..15; lanes 16-31 -> N=lane-16, K 16..31
// C/D (16x16 f32): lanes 0-15 -> N=lane, M=r; lanes 16-31 -> N=lane-16, M=8+r
DEV int a_koff(int lane) { return (lane & 16) ? 16 : 0; }
DEV int b_koff(int lane) { return (lane & 16) ? 16 : 0; }

DEV v16h load_a16(const _Float16* p) {
  v8h lo = *(const v8h*)p;
  v8h hi = *(const v8h*)(p + 8);
  return __builtin_shufflevector(lo, hi, 0, 1, 2, 3, 4, 5, 6, 7,
                                 8, 9, 10, 11, 12, 13, 14, 15);
}
DEV v16h load_b16(const _Float16* p) { return *(const v16h*)p; }

DEV v8f wmma16(v16h a, v16h b, v8f c) {
  return __builtin_amdgcn_wmma_f32_16x16x32_f16(false, a, false, b, (short)0, c, false, false);
}

// -------- weight convert+transpose: W f32 [nMat][K][Ncol] -> Wt f16 [nMat][Ncol][Kpad] --------
__global__ void k_wT(const float* __restrict__ W, _Float16* __restrict__ Wt,
                     int K, int Kpad, int Ncol, int total) {
  int idx = blockIdx.x * blockDim.x + threadIdx.x;
  if (idx >= total) return;
  int per = Ncol * Kpad;
  int m = idx / per;
  int r = idx - m * per;
  int col = r / Kpad, k = r - col * Kpad;
  Wt[idx] = (_Float16)((k < K) ? W[((size_t)m * K + k) * Ncol + col] : 0.f);
}

// ---------------- static structure: matrix0 (f32+f16), per-head block masks ----------------
__global__ void k_static(const float* __restrict__ adj, const float* __restrict__ dis,
                         const int* __restrict__ ring, const int* __restrict__ arom,
                         const int* __restrict__ alip, const unsigned char* __restrict__ hete,
                         float* __restrict__ mat, _Float16* __restrict__ mat16,
                         unsigned char* __restrict__ maskb) {
  int idx = blockIdx.x * blockDim.x + threadIdx.x;
  if (idx >= BG * LLn * LLn) return;
  int b = idx / (LLn * LLn);
  int r = idx - b * (LLn * LLn);
  int i = r / LLn, j = r - (r / LLn) * LLn;
  float a = adj[idx], dd = dis[idx];
  float inv = (dd == 0.f) ? dd : 1.f / dd;
  float mv = (a == 0.f) ? inv : a;
  mat[idx] = mv;
  mat16[idx] = (_Float16)mv;
  bool local = (a >= 0.8f) && (a != 0.825f);
  bool coj   = (a > 0.8f);
  bool het   = hete[b * LLn + j] && hete[b * LLn + i];
  int ri = ring[b * LLn + i], rj = ring[b * LLn + j];
  int ai = arom[b * LLn + i], aj = arom[b * LLn + j];
  int li = alip[b * LLn + i], lj = alip[b * LLn + j];
  unsigned m = 0;
  if (!local)                    m |= 1u;
  if (!coj)                      m |= 2u;
  if (!het)                      m |= 4u;
  if (!((rj == ri) && (rj > 0))) m |= 8u;
  if (!((aj == ai) && (aj > 0))) m |= 16u;
  if (!((lj == li) && (lj > 0))) m |= 32u;
  maskb[idx] = (unsigned char)m;
}

// ---------------- dense symmetric edge-feature scatter ----------------
__global__ void k_efp(const float* __restrict__ edge_attr, const int* __restrict__ ei,
                      float* __restrict__ efp, int nE) {
  int e = blockIdx.x * blockDim.x + threadIdx.x;
  if (e >= nE) return;
  int s = ei[e], d = ei[nE + e];
  int g = s / LLn;
  int u = s - g * LLn, v = d - g * LLn;
  for (int c = 0; c < EDIM; ++c) {
    float val = edge_attr[e * EDIM + c];
    efp[((size_t)(g * LLn + u) * LLn + v) * EDIM + c] = val;
    efp[((size_t)(g * LLn + v) * LLn + u) * EDIM + c] = val;
  }
}

// ---------------- GraphConv neighbor aggregation (segment_sum) ----------------
__global__ void k_agg(const float* __restrict__ xin, const int* __restrict__ ei,
                      float* __restrict__ agg, int kin, int nE) {
  int e = blockIdx.x;
  int t = threadIdx.x;
  if (t >= kin) return;
  int s = ei[e], d = ei[nE + e];
  atomicAdd(&agg[(size_t)d * kin + t], xin[(size_t)s * kin + t]);
}

// -------- GraphConv GEMM: relu(LN(agg@Wrel + b + x@Wroot)); f32 + f16 outputs --------
template <int KPAD>
__global__ void k_gconv(const float* __restrict__ xin, const _Float16* __restrict__ xin16,
                        const float* __restrict__ agg,
                        const _Float16* __restrict__ wrelT, const float* __restrict__ brel,
                        const _Float16* __restrict__ wrootT,
                        const float* __restrict__ lng, const float* __restrict__ lnb,
                        float* __restrict__ xout, _Float16* __restrict__ xout16, int kin) {
  __shared__ __align__(32) _Float16 sAgg[16 * KPAD];
  __shared__ __align__(32) _Float16 sX[16 * KPAD];
  __shared__ float sOut[16 * 128];
  __shared__ float sMV[32];
  int tid = threadIdx.x, lane = tid & 31, wave = tid >> 5;
  int row0 = blockIdx.x * 16;
  for (int i = tid; i < 16 * KPAD; i += 256) {
    int c = i - (i / KPAD) * KPAD;
    int r = i / KPAD;
    sAgg[i] = (_Float16)((c < kin) ? agg[(size_t)(row0 + r) * kin + c] : 0.f);
  }
  if (xin16) {  // f16 fast path (KPAD == kin == 128)
    const uint4* src = (const uint4*)(xin16 + (size_t)row0 * KPAD);
    uint4* dst = (uint4*)sX;
    for (int i = tid; i < 16 * KPAD / 8; i += 256) dst[i] = src[i];
  } else {
    for (int i = tid; i < 16 * KPAD; i += 256) {
      int c = i - (i / KPAD) * KPAD;
      int r = i / KPAD;
      sX[i] = (_Float16)((c < kin) ? xin[(size_t)(row0 + r) * kin + c] : 0.f);
    }
  }
  __syncthreads();
  int l15 = lane & 15, rowb = (lane & 16) ? 8 : 0;
  int ka = a_koff(lane), kb = b_koff(lane);
  int colb = wave * 16;
  v8f acc = {};
#pragma unroll
  for (int k0 = 0; k0 < KPAD; k0 += 32) {
    v16h a1 = load_a16(&sAgg[l15 * KPAD + k0 + ka]);
    v16h b1 = load_b16(&wrelT[(size_t)(colb + l15) * KPAD + k0 + kb]);
    acc = wmma16(a1, b1, acc);
    v16h a2 = load_a16(&sX[l15 * KPAD + k0 + ka]);
    v16h b2 = load_b16(&wrootT[(size_t)(colb + l15) * KPAD + k0 + kb]);
    acc = wmma16(a2, b2, acc);
  }
  float bb = brel[colb + l15];
#pragma unroll
  for (int r = 0; r < 8; ++r)
    sOut[(rowb + r) * 128 + colb + l15] = acc[r] + bb;
  __syncthreads();
  if (tid < 16) {
    float s = 0.f, s2 = 0.f;
    for (int c = 0; c < 128; ++c) { float v = sOut[tid * 128 + c]; s += v; s2 += v * v; }
    float m = s * (1.f / 128.f);
    sMV[tid * 2] = m; sMV[tid * 2 + 1] = s2 * (1.f / 128.f) - m * m;
  }
  __syncthreads();
#pragma unroll
  for (int r = 0; r < 8; ++r) {
    int row = rowb + r, col = colb + l15;
    float m = sMV[row * 2], var = sMV[row * 2 + 1];
    float o = (sOut[row * 128 + col] - m) * rsqrtf(var + 1e-5f) * lng[col] + lnb[col];
    o = fmaxf(o, 0.f);
    xout[(size_t)(row0 + row) * DD + col] = o;
    xout16[(size_t)(row0 + row) * DD + col] = (_Float16)o;
  }
}

// -------- QKV + wv2 projections; Q/K row-major f16, V and v2 transposed [d][l] f16 --------
__global__ void k_qkv(const _Float16* __restrict__ xin16,
                      const _Float16* __restrict__ wqT, const _Float16* __restrict__ wkT,
                      const _Float16* __restrict__ wvT, const _Float16* __restrict__ wv2T,
                      _Float16* __restrict__ Qh, _Float16* __restrict__ Kh,
                      _Float16* __restrict__ Vt, _Float16* __restrict__ v2t) {
  __shared__ __align__(32) _Float16 sX[16 * 128];
  int tid = threadIdx.x, lane = tid & 31, wave = tid >> 5;
  int row0 = blockIdx.x * 16;
  int b = row0 / LLn, l0 = row0 - b * LLn;
  {
    const uint4* src = (const uint4*)(xin16 + (size_t)row0 * 128);
    uint4* dst = (uint4*)sX;
    for (int i = tid; i < 16 * 128 / 8; i += 256) dst[i] = src[i];
  }
  __syncthreads();
  int l15 = lane & 15, rowb = (lane & 16) ? 8 : 0;
  int ka = a_koff(lane), kb = b_koff(lane);
  // A fragments are invariant across column tiles: load once.
  v16h a0 = load_a16(&sX[l15 * 128 + 0 + ka]);
  v16h a1 = load_a16(&sX[l15 * 128 + 32 + ka]);
  v16h a2 = load_a16(&sX[l15 * 128 + 64 + ka]);
  v16h a3 = load_a16(&sX[l15 * 128 + 96 + ka]);
  for (int ct = wave; ct < 100; ct += 8) {
    const _Float16* Wt;
    int colb;
    if (ct < 32)      { Wt = wqT;  colb = ct * 16; }
    else if (ct < 64) { Wt = wkT;  colb = (ct - 32) * 16; }
    else if (ct < 96) { Wt = wvT;  colb = (ct - 64) * 16; }
    else              { Wt = wv2T; colb = (ct - 96) * 16; }
    int col = colb + l15;
    const _Float16* wp = Wt + (size_t)col * 128 + kb;
    v8f acc = {};
    acc = wmma16(a0, load_b16(wp), acc);
    acc = wmma16(a1, load_b16(wp + 32), acc);
    acc = wmma16(a2, load_b16(wp + 64), acc);
    acc = wmma16(a3, load_b16(wp + 96), acc);
    if (ct < 64) {
      _Float16* O = (ct < 32) ? Qh : Kh;
#pragma unroll
      for (int r = 0; r < 8; ++r)
        O[(size_t)(row0 + rowb + r) * HD + col] = (_Float16)acc[r];
    } else if (ct < 96) {
      int h = col >> 6, dd = col & 63;
#pragma unroll
      for (int r = 0; r < 8; ++r)
        Vt[((size_t)(b * HHn + h) * DKC + dd) * LLn + l0 + rowb + r] = (_Float16)acc[r];
    } else {
#pragma unroll
      for (int r = 0; r < 8; ++r)
        v2t[((size_t)b * DKC + col) * LLn + l0 + rowb + r] = (_Float16)acc[r];
    }
  }
}

// ---------------- edge-bias MLP: tanh(efp@W1+b1)@W2+b2 ----------------
__global__ void k_bias(const float* __restrict__ efp,
                       const float* __restrict__ w1, const float* __restrict__ b1,
                       const float* __restrict__ w2, const float* __restrict__ b2,
                       float* __restrict__ bias) {
  int idx = blockIdx.x * blockDim.x + threadIdx.x;
  if (idx >= BG * LLn * LLn) return;
  float e[EDIM];
#pragma unroll
  for (int c = 0; c < EDIM; ++c) e[c] = efp[(size_t)idx * EDIM + c];
  float out = b2[0];
  for (int j = 0; j < DKC; ++j) {
    float t = b1[j];
#pragma unroll
    for (int c = 0; c < EDIM; ++c) t += e[c] * w1[c * DKC + j];
    out += tanhf(t) * w2[j];
  }
  bias[idx] = out;
}

// ---------------- scores = Q.K^T/8 + bias, masked, per (b,h) ----------------
__global__ void k_scores(const _Float16* __restrict__ Qh, const _Float16* __restrict__ Kh,
                         const float* __restrict__ bias, const unsigned char* __restrict__ maskb,
                         float* __restrict__ scores) {
  int bh = blockIdx.x;
  int b = bh / HHn, h = bh - (bh / HHn) * HHn;
  int tid = threadIdx.x, lane = tid & 31, wave = tid >> 5;
  int l15 = lane & 15, rowb = (lane & 16) ? 8 : 0;
  int ka = a_koff(lane), kb = b_koff(lane);
  const _Float16* Qb = Qh + (size_t)b * LLn * HD + h * DKC;
  const _Float16* Kb = Kh + (size_t)b * LLn * HD + h * DKC;
  for (int t = wave; t < 64; t += 8) {
    int mbase = (t >> 3) * 16, nbase = (t & 7) * 16;
    v8f acc = {};
#pragma unroll
    for (int k0 = 0; k0 < DKC; k0 += 32) {
      v16h a  = load_a16(Qb + (size_t)(mbase + l15) * HD + k0 + ka);
      v16h bv = load_b16(Kb + (size_t)(nbase + l15) * HD + k0 + kb);
      acc = wmma16(a, bv, acc);
    }
    int n = nbase + l15;
#pragma unroll
    for (int r = 0; r < 8; ++r) {
      int m = mbase + rowb + r;
      int ij = (b * LLn + m) * LLn + n;
      float s = acc[r] * 0.125f + bias[ij];
      if (h >= 2 && ((maskb[ij] >> (h - 2)) & 1)) s = -1e9f;
      scores[((size_t)(b * HHn + h) * LLn + m) * LLn + n] = s;
    }
  }
}

// ---------------- row softmax (128 elems), in place; also emit f16 copy ----------------
__global__ void k_softmax(float* __restrict__ scores, _Float16* __restrict__ attn16) {
  __shared__ float red[128];
  int row = blockIdx.x, t = threadIdx.x;
  float* p = scores + (size_t)row * 128;
  float v = p[t];
  red[t] = v;
  __syncthreads();
  for (int s = 64; s > 0; s >>= 1) { if (t < s) red[t] = fmaxf(red[t], red[t + s]); __syncthreads(); }
  float mx = red[0];
  __syncthreads();
  float ev = __expf(v - mx);
  red[t] = ev;
  __syncthreads();
  for (int s = 64; s > 0; s >>= 1) { if (t < s) red[t] += red[t + s]; __syncthreads(); }
  float a = ev / red[0];
  p[t] = a;
  attn16[(size_t)row * 128 + t] = (_Float16)a;
}

// ---------------- context = attn @ V per (b,h) -> ctx[:, h*64+d] ----------------
__global__ void k_context(const _Float16* __restrict__ attn16, const _Float16* __restrict__ Vt,
                          _Float16* __restrict__ ctx) {
  int bh = blockIdx.x;
  int b = bh / HHn, h = bh - (bh / HHn) * HHn;
  int tid = threadIdx.x, lane = tid & 31, wave = tid >> 5;
  int l15 = lane & 15, rowb = (lane & 16) ? 8 : 0;
  int ka = a_koff(lane), kb = b_koff(lane);
  const _Float16* Ab = attn16 + (size_t)(b * HHn + h) * LLn * LLn;
  const _Float16* Vb = Vt + (size_t)(b * HHn + h) * DKC * LLn;
  for (int t = wave; t < 32; t += 8) {
    int mbase = (t >> 2) * 16, nbase = (t & 3) * 16;
    v8f acc = {};
#pragma unroll
    for (int k0 = 0; k0 < LLn; k0 += 32) {
      v16h a  = load_a16(Ab + (size_t)(mbase + l15) * LLn + k0 + ka);
      v16h bv = load_b16(Vb + (size_t)(nbase + l15) * LLn + k0 + kb);
      acc = wmma16(a, bv, acc);
    }
#pragma unroll
    for (int r = 0; r < 8; ++r)
      ctx[(size_t)(b * LLn + mbase + rowb + r) * FCK + h * DKC + nbase + l15] = (_Float16)acc[r];
  }
}

// ---------------- context2 = matrix @ (x@wv2) per b -> ctx[:, 512+d] ----------------
__global__ void k_context2(const _Float16* __restrict__ mat16, const _Float16* __restrict__ v2t,
                           _Float16* __restrict__ ctx) {
  int b = blockIdx.x;
  int tid = threadIdx.x, lane = tid & 31, wave = tid >> 5;
  int l15 = lane & 15, rowb = (lane & 16) ? 8 : 0;
  int ka = a_koff(lane), kb = b_koff(lane);
  const _Float16* Mb = mat16 + (size_t)b * LLn * LLn;
  const _Float16* Vb = v2t + (size_t)b * DKC * LLn;
  for (int t = wave; t < 32; t += 8) {
    int mbase = (t >> 2) * 16, nbase = (t & 3) * 16;
    v8f acc = {};
#pragma unroll
    for (int k0 = 0; k0 < LLn; k0 += 32) {
      v16h a  = load_a16(Mb + (size_t)(mbase + l15) * LLn + k0 + ka);
      v16h bv = load_b16(Vb + (size_t)(nbase + l15) * LLn + k0 + kb);
      acc = wmma16(a, bv, acc);
    }
#pragma unroll
    for (int r = 0; r < 8; ++r)
      ctx[(size_t)(b * LLn + mbase + rowb + r) * FCK + HD + nbase + l15] = (_Float16)acc[r];
  }
}

// ---------------- fusion gate: LN(9) -> 6 -> relu -> 1 -> sigmoid; matrix *= gate ----------------
__global__ void k_gate(const float* __restrict__ matOld, const float* __restrict__ attn,
                       const float* __restrict__ lng, const float* __restrict__ lnb,
                       const float* __restrict__ w1, const float* __restrict__ b1,
                       const float* __restrict__ w2, const float* __restrict__ b2,
                       float* __restrict__ matNew, _Float16* __restrict__ matNew16) {
  int idx = blockIdx.x * blockDim.x + threadIdx.x;
  if (idx >= BG * LLn * LLn) return;
  int b = idx / (LLn * LLn);
  int r = idx - b * (LLn * LLn);
  int i = r / LLn, j = r - (r / LLn) * LLn;
  float ch[HHn + 1];
  ch[0] = matOld[(size_t)(b * LLn + j) * LLn + i];
#pragma unroll
  for (int h = 0; h < HHn; ++h)
    ch[1 + h] = attn[((size_t)(b * HHn + h) * LLn + j) * LLn + i];
  float s = 0.f, s2 = 0.f;
#pragma unroll
  for (int c = 0; c < HHn + 1; ++c) { s += ch[c]; s2 += ch[c] * ch[c]; }
  float m = s * (1.f / 9.f);
  float rstd = rsqrtf(s2 * (1.f / 9.f) - m * m + 1e-5f);
  float tv[HHn + 1];
#pragma unroll
  for (int c = 0; c < HHn + 1; ++c) tv[c] = (ch[c] - m) * rstd * lng[c] + lnb[c];
  float g = b2[0];
#pragma unroll
  for (int k = 0; k < 6; ++k) {
    float u = b1[k];
#pragma unroll
    for (int c = 0; c < HHn + 1; ++c) u += tv[c] * w1[c * 6 + k];
    g += fmaxf(u, 0.f) * w2[k];
  }
  float mv = matOld[idx] * (1.f / (1.f + __expf(-g)));
  matNew[idx] = mv;
  matNew16[idx] = (_Float16)mv;
}

// ---- fc GEMM (K=576) + LN -> sigmoid -> +xb -> LN; f32 + f16 outputs ----
__global__ void k_fcout(const _Float16* __restrict__ ctx, const _Float16* __restrict__ fcT,
                        const float* __restrict__ mlg, const float* __restrict__ mlb,
                        const float* __restrict__ elg, const float* __restrict__ elb,
                        const float* __restrict__ xb,
                        float* __restrict__ xout, _Float16* __restrict__ xout16) {
  __shared__ __align__(32) _Float16 sA[16 * FCK];
  __shared__ float sOut[16 * 128];
  __shared__ float sMV[32];
  int tid = threadIdx.x, lane = tid & 31, wave = tid >> 5;
  int row0 = blockIdx.x * 16;
  {
    const uint4* src = (const uint4*)(ctx + (size_t)row0 * FCK);
    uint4* dst = (uint4*)sA;
    for (int i = tid; i < 16 * FCK / 8; i += 256) dst[i] = src[i];
  }
  __syncthreads();
  int l15 = lane & 15, rowb = (lane & 16) ? 8 : 0;
  int ka = a_koff(lane), kb = b_koff(lane);
  int colb = wave * 16;
  v8f acc = {};
#pragma unroll
  for (int k0 = 0; k0 < FCK; k0 += 32) {
    v16h a  = load_a16(&sA[l15 * FCK + k0 + ka]);
    v16h bv = load_b16(&fcT[(size_t)(colb + l15) * FCK + k0 + kb]);
    acc = wmma16(a, bv, acc);
  }
#pragma unroll
  for (int r = 0; r < 8; ++r)
    sOut[(rowb + r) * 128 + colb + l15] = acc[r];
  __syncthreads();
  if (tid < 16) {
    float s = 0.f, s2 = 0.f;
    for (int c = 0; c < 128; ++c) { float v = sOut[tid * 128 + c]; s += v; s2 += v * v; }
    float m = s * (1.f / 128.f);
    sMV[tid * 2] = m; sMV[tid * 2 + 1] = s2 * (1.f / 128.f) - m * m;
  }
  __syncthreads();
#pragma unroll
  for (int r = 0; r < 8; ++r) {
    int row = rowb + r, col = colb + l15;
    float m = sMV[row * 2], var = sMV[row * 2 + 1];
    float o = (sOut[row * 128 + col] - m) * rsqrtf(var + 1e-5f) * mlg[col] + mlb[col];
    float sig = 1.f / (1.f + __expf(-o));
    sOut[row * 128 + col] = sig + xb[(size_t)(row0 + row) * DD + col];
  }
  __syncthreads();
  if (tid < 16) {
    float s = 0.f, s2 = 0.f;
    for (int c = 0; c < 128; ++c) { float v = sOut[tid * 128 + c]; s += v; s2 += v * v; }
    float m = s * (1.f / 128.f);
    sMV[tid * 2] = m; sMV[tid * 2 + 1] = s2 * (1.f / 128.f) - m * m;
  }
  __syncthreads();
#pragma unroll
  for (int r = 0; r < 8; ++r) {
    int row = rowb + r, col = colb + l15;
    float m = sMV[row * 2], var = sMV[row * 2 + 1];
    float o = (sOut[row * 128 + col] - m) * rsqrtf(var + 1e-5f) * elg[col] + elb[col];
    xout[(size_t)(row0 + row) * DD + col] = o;
    xout16[(size_t)(row0 + row) * DD + col] = (_Float16)o;
  }
}

extern "C" void kernel_launch(void* const* d_in, const int* in_sizes, int n_in,
                              void* d_out, int out_size, void* d_ws, size_t ws_size,
                              hipStream_t stream) {
  (void)in_sizes; (void)n_in; (void)out_size; (void)ws_size;
  const float* x         = (const float*)d_in[0];
  const float* edge_attr = (const float*)d_in[1];
  const float* adj       = (const float*)d_in[2];
  const float* dis       = (const float*)d_in[3];
  const float* eb1_b     = (const float*)d_in[4];
  const float* eb1_w     = (const float*)d_in[5];
  const float* eb2_b     = (const float*)d_in[6];
  const float* eb2_w     = (const float*)d_in[7];
  const float* enc_ln_b  = (const float*)d_in[8];
  const float* enc_ln_g  = (const float*)d_in[9];
  const float* fcp       = (const float*)d_in[10];
  const float* fu1_b     = (const float*)d_in[11];
  const float* fu1_w     = (const float*)d_in[12];
  const float* fu2_b     = (const float*)d_in[13];
  const float* fu2_w     = (const float*)d_in[14];
  const float* fu_ln_b   = (const float*)d_in[15];
  const float* fu_ln_g   = (const float*)d_in[16];
  const float* g0_rel_b  = (const float*)d_in[17];
  const float* g0_rel_w  = (const float*)d_in[18];
  const float* g0_root_w = (const float*)d_in[19];
  const float* g_rel_b   = (const float*)d_in[20];
  const float* g_rel_w   = (const float*)d_in[21];
  const float* g_root_w  = (const float*)d_in[22];
  const float* ln_b      = (const float*)d_in[23];
  const float* ln_g      = (const float*)d_in[24];
  const float* mha_ln_b  = (const float*)d_in[25];
  const float* mha_ln_g  = (const float*)d_in[26];
  const float* wk        = (const float*)d_in[27];
  const float* wq        = (const float*)d_in[28];
  const float* wv        = (const float*)d_in[29];
  const float* wv2       = (const float*)d_in[30];
  const int*   edge_index= (const int*)d_in[31];
  /* d_in[32] = batch (unused; graphs contiguous) */
  const int*   ring      = (const int*)d_in[33];
  const int*   arom      = (const int*)d_in[34];
  const int*   alip      = (const int*)d_in[35];
  const unsigned char* hete = (const unsigned char*)d_in[36];
  const int nE = BG * EPG;
  const int nLL = BG * LLn * LLn;

  size_t off = 0;
  auto alloc = [&](size_t bytes) -> void* {
    void* p = (char*)d_ws + off;
    off += (bytes + 255) & ~(size_t)255;
    return p;
  };
  float*    agg    = (float*)alloc((size_t)NN * DD * 4);
  float*    bufG   = (float*)alloc((size_t)NN * DD * 4);   // gconv output (xb), f32
  _Float16* bufG16 = (_Float16*)alloc((size_t)NN * DD * 2);
  float*    bufE   = (float*)alloc((size_t)NN * DD * 4);   // encoder output, f32
  _Float16* bufE16 = (_Float16*)alloc((size_t)NN * DD * 2);
  _Float16* Qh     = (_Float16*)alloc((size_t)NN * HD * 2);
  _Float16* Kh     = (_Float16*)alloc((size_t)NN * HD * 2);
  _Float16* Vt     = (_Float16*)alloc((size_t)NN * HD * 2);   // [b][h][d][l]
  _Float16* v2t    = (_Float16*)alloc((size_t)NN * DKC * 2);  // [b][d][l]
  float*    scores = (float*)alloc((size_t)BG * HHn * LLn * LLn * 4);
  _Float16* attn16 = (_Float16*)alloc((size_t)BG * HHn * LLn * LLn * 2);
  float*    biasb  = (float*)alloc((size_t)nLL * 4);
  float*    matA   = (float*)alloc((size_t)nLL * 4);
  float*    matB   = (float*)alloc((size_t)nLL * 4);
  _Float16* mat16A = (_Float16*)alloc((size_t)nLL * 2);
  _Float16* mat16B = (_Float16*)alloc((size_t)nLL * 2);
  float*    efp    = (float*)alloc((size_t)nLL * EDIM * 4);
  unsigned char* maskb = (unsigned char*)alloc((size_t)nLL);
  _Float16* ctx    = (_Float16*)alloc((size_t)NN * FCK * 2);
  _Float16* wqT    = (_Float16*)alloc((size_t)NLAY * HD * DD * 2);
  _Float16* wkT    = (_Float16*)alloc((size_t)NLAY * HD * DD * 2);
  _Float16* wvT    = (_Float16*)alloc((size_t)NLAY * HD * DD * 2);
  _Float16* wv2T   = (_Float16*)alloc((size_t)NLAY * DKC * DD * 2);
  _Float16* fcT    = (_Float16*)alloc((size_t)NLAY * DD * FCK * 2);
  _Float16* g0relT = (_Float16*)alloc((size_t)DD * 64 * 2);
  _Float16* g0rootT= (_Float16*)alloc((size_t)DD * 64 * 2);
  _Float16* grelT  = (_Float16*)alloc((size_t)(NLAY - 1) * DD * DD * 2);
  _Float16* grootT = (_Float16*)alloc((size_t)(NLAY - 1) * DD * DD * 2);

  // ---- one-time conversions ----
  {
    int t;
    t = NLAY * HD * DD;
    k_wT<<<(t + 255) / 256, 256, 0, stream>>>(wq, wqT, DD, DD, HD, t);
    k_wT<<<(t + 255) / 256, 256, 0, stream>>>(wk, wkT, DD, DD, HD, t);
    k_wT<<<(t + 255) / 256, 256, 0, stream>>>(wv, wvT, DD, DD, HD, t);
    t = NLAY * DKC * DD;
    k_wT<<<(t + 255) / 256, 256, 0, stream>>>(wv2, wv2T, DD, DD, DKC, t);
    t = NLAY * DD * FCK;
    k_wT<<<(t + 255) / 256, 256, 0, stream>>>(fcp, fcT, FCK, FCK, DD, t);
    t = DD * 64;
    k_wT<<<(t + 255) / 256, 256, 0, stream>>>(g0_rel_w, g0relT, IN_DIM, 64, DD, t);
    k_wT<<<(t + 255) / 256, 256, 0, stream>>>(g0_root_w, g0rootT, IN_DIM, 64, DD, t);
    t = (NLAY - 1) * DD * DD;
    k_wT<<<(t + 255) / 256, 256, 0, stream>>>(g_rel_w, grelT, DD, DD, DD, t);
    k_wT<<<(t + 255) / 256, 256, 0, stream>>>(g_root_w, grootT, DD, DD, DD, t);
  }

  k_static<<<(nLL + 255) / 256, 256, 0, stream>>>(adj, dis, ring, arom, alip, hete,
                                                  matA, mat16A, maskb);
  hipMemsetAsync(efp, 0, (size_t)nLL * EDIM * 4, stream);
  k_efp<<<(nE + 255) / 256, 256, 0, stream>>>(edge_attr, edge_index, efp, nE);

  float*    matOld = matA;   float*    matNew = matB;
  _Float16* m16Old = mat16A; _Float16* m16New = mat16B;
  for (int i = 0; i < NLAY; ++i) {
    const float* xin = (i == 0) ? x : bufE;
    int kin = (i == 0) ? IN_DIM : DD;
    const float* brel = (i == 0) ? g0_rel_b : g_rel_b + (size_t)(i - 1) * DD;

    hipMemsetAsync(agg, 0, (size_t)NN * kin * 4, stream);
    k_agg<<<nE, 128, 0, stream>>>(xin, edge_index, agg, kin, nE);
    if (i == 0)
      k_gconv<64><<<NN / 16, 256, 0, stream>>>(xin, nullptr, agg, g0relT, brel, g0rootT,
                                               ln_g, ln_b, bufG, bufG16, kin);
    else
      k_gconv<128><<<NN / 16, 256, 0, stream>>>(xin, bufE16, agg,
                                                grelT + (size_t)(i - 1) * DD * DD, brel,
                                                grootT + (size_t)(i - 1) * DD * DD,
                                                ln_g + (size_t)i * DD, ln_b + (size_t)i * DD,
                                                bufG, bufG16, kin);
    k_qkv<<<NN / 16, 256, 0, stream>>>(bufG16,
                                       wqT + (size_t)i * HD * DD, wkT + (size_t)i * HD * DD,
                                       wvT + (size_t)i * HD * DD, wv2T + (size_t)i * DKC * DD,
                                       Qh, Kh, Vt, v2t);
    k_bias<<<(nLL + 255) / 256, 256, 0, stream>>>(efp,
                                                  eb1_w + (size_t)i * EDIM * DKC,
                                                  eb1_b + (size_t)i * DKC,
                                                  eb2_w + (size_t)i * DKC, eb2_b + i, biasb);
    k_scores<<<BG * HHn, 256, 0, stream>>>(Qh, Kh, biasb, maskb, scores);
    k_softmax<<<BG * HHn * LLn, 128, 0, stream>>>(scores, attn16);
    k_context<<<BG * HHn, 256, 0, stream>>>(attn16, Vt, ctx);
    k_context2<<<BG, 256, 0, stream>>>(m16Old, v2t, ctx);
    k_gate<<<(nLL + 255) / 256, 256, 0, stream>>>(matOld, scores,
                                                  fu_ln_g + (size_t)i * 9, fu_ln_b + (size_t)i * 9,
                                                  fu1_w + (size_t)i * 54, fu1_b + (size_t)i * 6,
                                                  fu2_w + (size_t)i * 6, fu2_b + i,
                                                  matNew, m16New);
    k_fcout<<<NN / 16, 256, 0, stream>>>(ctx, fcT + (size_t)i * DD * FCK,
                                         mha_ln_g + (size_t)i * DD, mha_ln_b + (size_t)i * DD,
                                         enc_ln_g + (size_t)i * DD, enc_ln_b + (size_t)i * DD,
                                         bufG, bufE, bufE16);
    float* tf = matOld; matOld = matNew; matNew = tf;
    _Float16* th = m16Old; m16Old = m16New; m16New = th;
  }
  hipMemcpyAsync(d_out, bufE, (size_t)NN * DD * 4, hipMemcpyDeviceToDevice, stream);
}
